// MultiHeadAttention_44074954392185
// MI455X (gfx1250) — compile-verified, matrix operand order checked by reading, not run on hardware
//
#include <hip/hip_runtime.h>
#include <hip/hip_bf16.h>

#define BDIM 4
#define TDIM 2048
#define CDIM 1024
#define HDIM 16
#define DDIM 64

typedef __bf16 bf16_t;
typedef __attribute__((ext_vector_type(16))) __bf16 v16bf;
typedef __attribute__((ext_vector_type(8)))  float  v8f;

#if defined(__HIP_DEVICE_COMPILE__) && __has_builtin(__builtin_amdgcn_global_load_async_to_lds_b128)
#define HAVE_ASYNC_LDS 1
#endif

#ifdef HAVE_ASYNC_LDS
typedef int v4i_ __attribute__((vector_size(16)));
typedef __attribute__((address_space(1))) v4i_* as1_v4i;
typedef __attribute__((address_space(3))) v4i_* as3_v4i;
#endif

static __device__ __forceinline__ v16bf ld_frag(const bf16_t* p) {
    return *(const v16bf*)(p);
}

static __device__ __forceinline__ v8f wmma_bf16(v16bf a, v16bf b, v8f c) {
    // (neg_a, A, neg_b, B, c_mod, C, reuse_a, reuse_b)
    return __builtin_amdgcn_wmma_f32_16x16x32_bf16(false, a, false, b, (short)0, c,
                                                   false, false);
}

// 16-byte global -> LDS copy; async (ASYNCcnt) when the builtin exists.
static __device__ __forceinline__ void g2l_copy16(const bf16_t* g, bf16_t* l) {
#ifdef HAVE_ASYNC_LDS
    __builtin_amdgcn_global_load_async_to_lds_b128((as1_v4i)(g), (as3_v4i)(l), 0, 0);
#else
    *(uint4*)l = *(const uint4*)g;
#endif
}

// Wait until at most N async group-ops remain in flight (s_wait_asynccnt N).
template <int N>
static __device__ __forceinline__ void g2l_wait() {
#if defined(HAVE_ASYNC_LDS) && __has_builtin(__builtin_amdgcn_s_wait_asynccnt)
    __builtin_amdgcn_s_wait_asynccnt(N);
#endif
}

// ---------------------------------------------------------------------------
// fp32 -> bf16 elementwise convert (vectorized x4)
// ---------------------------------------------------------------------------
__global__ void __launch_bounds__(256)
k_f32_to_bf16(const float* __restrict__ s, bf16_t* __restrict__ d, int n) {
    int i = (blockIdx.x * blockDim.x + threadIdx.x) * 4;
    if (i + 3 < n) {
        float4 v = *(const float4*)(s + i);
        d[i + 0] = (bf16_t)v.x;
        d[i + 1] = (bf16_t)v.y;
        d[i + 2] = (bf16_t)v.z;
        d[i + 3] = (bf16_t)v.w;
    } else {
        for (; i < n; ++i) d[i] = (bf16_t)s[i];
    }
}

// ---------------------------------------------------------------------------
// Fused QKV projection + RoPE + RMSNorm epilogue.
// Block: 256 threads (8 waves). Block tile: 256 x 64 (one head), wave 32 x 64.
// Two-phase (ping/pong) async pipeline with compile-time buffer indices so
// LDS addresses fold into ds_load offsets and accumulators coalesce in-place.
// grid = (8192/256, 16 heads, 3 {q,k,v}); output [b][h][t][d] bf16.
// ---------------------------------------------------------------------------
__global__ void __launch_bounds__(256, 2)
k_qkv(const bf16_t* __restrict__ xb,
      const bf16_t* __restrict__ wq,
      const bf16_t* __restrict__ wk,
      const bf16_t* __restrict__ wv,
      const float*  __restrict__ cosp,
      const float*  __restrict__ sinp,
      bf16_t* __restrict__ qb, bf16_t* __restrict__ kb, bf16_t* __restrict__ vb) {
    __shared__ alignas(32) bf16_t Alds[2][256][32];
    __shared__ alignas(32) bf16_t Blds[2][64][32];

    const int m0 = blockIdx.x * 256;
    const int h  = blockIdx.y;
    const int z  = blockIdx.z;
    const int n0 = h * DDIM;
    const bf16_t* W   = (z == 0) ? wq : (z == 1) ? wk : wv;
    bf16_t*       OUT = (z == 0) ? qb : (z == 1) ? kb : vb;

    const int tid  = threadIdx.x;
    const int wave = tid >> 5;
    const int lane = tid & 31;
    const int ln   = lane & 15;   // N index within fragment
    const int lh   = lane >> 4;   // lane half

    v8f acc[2][4] = {};

    auto stage = [&](int buf, int k0) {
#pragma unroll
        for (int q = 0; q < 4; ++q) {           // A: 1024 b128 chunks / 256 thr
            int chunk = tid + q * 256;
            int row = chunk >> 2, off = (chunk & 3) * 8;
            g2l_copy16(&xb[(size_t)(m0 + row) * CDIM + k0 + off],
                       &Alds[buf][row][off]);
        }
        int brow = tid >> 2, boff = (tid & 3) * 8;  // B: 256 b128 chunks
        g2l_copy16(&W[(size_t)(n0 + brow) * CDIM + k0 + boff],
                   &Blds[buf][brow][boff]);
    };

    auto kstep = [&](int buf, int k0n, bool hn) {  // buf is a literal at call
        if (hn) {
            stage(buf ^ 1, k0n);   // next tile in flight during this WMMA block
            g2l_wait<5>();         // only current tile's group must be done
        } else {
            g2l_wait<0>();
        }
        __syncthreads();
        v16bf a0 = ld_frag(&Alds[buf][wave * 32 + ln][lh * 16]);
        v16bf a1 = ld_frag(&Alds[buf][wave * 32 + 16 + ln][lh * 16]);
#pragma unroll
        for (int t = 0; t < 4; ++t) {
            v16bf b = ld_frag(&Blds[buf][t * 16 + ln][lh * 16]);
            acc[0][t] = wmma_bf16(a0, b, acc[0][t]);
            acc[1][t] = wmma_bf16(a1, b, acc[1][t]);
        }
        __syncthreads();
    };

    stage(0, 0);
    for (int it = 0; it < CDIM / 32; it += 2) {
        kstep(0, (it + 1) * 32, true);                        // it+1 < 32 always
        kstep(1, (it + 2) * 32, (it + 2) < (CDIM / 32));
    }

#pragma unroll
    for (int u = 0; u < 2; ++u) {
        const int rbase = m0 + wave * 32 + u * 16;
        if (z < 2) {
            // RoPE + RMSNorm over the full head (wave owns all 64 dims)
#pragma unroll
            for (int r = 0; r < 8; ++r) {
                int mrow = rbase + r + lh * 8;
                int t_ = mrow & (TDIM - 1);
                int b_ = mrow >> 11;
                float val[4], c[4], s[4];
#pragma unroll
                for (int t = 0; t < 4; ++t) {
                    int d = t * 16 + ln;
                    val[t] = acc[u][t][r];
                    c[t] = cosp[t_ * DDIM + d];
                    s[t] = sinp[t_ * DDIM + d];
                }
                // rotate_half: d<32 pairs with d+32 (neg); d>=32 with d-32 (pos)
                float rot[4];
                rot[0] = val[0] * c[0] - val[2] * s[0];
                rot[1] = val[1] * c[1] - val[3] * s[1];
                rot[2] = val[2] * c[2] + val[0] * s[2];
                rot[3] = val[3] * c[3] + val[1] * s[3];
                float ss = rot[0]*rot[0] + rot[1]*rot[1] + rot[2]*rot[2] + rot[3]*rot[3];
                ss += __shfl_xor(ss, 1, 32);
                ss += __shfl_xor(ss, 2, 32);
                ss += __shfl_xor(ss, 4, 32);
                ss += __shfl_xor(ss, 8, 32);
                float inv = rsqrtf(ss * (1.0f / DDIM) + 1e-6f);
                size_t base = (((size_t)b_ * HDIM + h) * TDIM + t_) * DDIM;
#pragma unroll
                for (int t = 0; t < 4; ++t)
                    OUT[base + t * 16 + ln] = (bf16_t)(rot[t] * inv);
            }
        } else {
#pragma unroll
            for (int r = 0; r < 8; ++r) {
                int mrow = rbase + r + lh * 8;
                int t_ = mrow & (TDIM - 1);
                int b_ = mrow >> 11;
                size_t base = (((size_t)b_ * HDIM + h) * TDIM + t_) * DDIM;
#pragma unroll
                for (int t = 0; t < 4; ++t)
                    OUT[base + t * 16 + ln] = (bf16_t)acc[u][t][r];
            }
        }
    }
}

// ---------------------------------------------------------------------------
// Flash attention per (b,h). Block: 128 threads (4 waves), 64 q-rows/block,
// 64-key j-blocks. Two-phase ping/pong: K async double-buffered, V staged
// (transposed) behind the P*V WMMAs, WGP-scope prefetch one tile ahead.
// grid = (T/64, B*H). 16 WMMA per wave per j-step.
// ---------------------------------------------------------------------------
__global__ void __launch_bounds__(128, 2)
k_attn(const bf16_t* __restrict__ qb, const bf16_t* __restrict__ kb,
       const bf16_t* __restrict__ vb, bf16_t* __restrict__ yb) {
    __shared__ alignas(32) bf16_t Klds[2][64][64];    // [key][d]
    __shared__ alignas(32) bf16_t VldsT[2][64][64];   // transposed: [d][key]
    __shared__ alignas(32) bf16_t Plds[4][16][64];    // per-wave P relayout

    const int bh = blockIdx.y;
    const int b_ = bh / HDIM;
    const int h  = bh % HDIM;
    const int q0 = blockIdx.x * 64;

    const int tid  = threadIdx.x;
    const int wave = tid >> 5;
    const int lane = tid & 31;
    const int ln   = lane & 15;
    const int lh   = lane >> 4;

    const size_t headbase = ((size_t)b_ * HDIM + h) * (size_t)TDIM * DDIM;

    auto stageK = [&](int buf, int j0) {   // 512 b128 chunks / 128 thr = 4 each
#pragma unroll
        for (int q = 0; q < 4; ++q) {
            int chunk = tid + q * 128;
            int key = chunk >> 3, off = (chunk & 7) * 8;
            g2l_copy16(&kb[headbase + (size_t)(j0 + key) * DDIM + off],
                       &Klds[buf][key][off]);
        }
    };
    auto stageV = [&](int buf, int j0) {   // sync transpose into [d][key]
        int key = tid >> 1, ch0 = (tid & 1) * 32;
#pragma unroll
        for (int half = 0; half < 2; ++half) {
            int ch = ch0 + half * 16;
            v16bf vv = *(const v16bf*)&vb[headbase + (size_t)(j0 + key) * DDIM + ch];
#pragma unroll
            for (int e = 0; e < 16; ++e) VldsT[buf][ch + e][key] = vv[e];
        }
    };

    // Q fragments stay in registers: 16 rows x 64 dims = 2 A-fragments
    const bf16_t* Qp = qb + headbase + (size_t)(q0 + wave * 16 + ln) * DDIM;
    v16bf qa0 = *(const v16bf*)(Qp + lh * 16);
    v16bf qa1 = *(const v16bf*)(Qp + 32 + lh * 16);

    v8f o[4] = {};
    float m_run[8], l_run[8];
#pragma unroll
    for (int r = 0; r < 8; ++r) { m_run[r] = -1e30f; l_run[r] = 0.0f; }

    auto jstep = [&](int buf, int j0n, bool hn) {   // buf is a literal at call
        if (hn) {
            stageK(buf ^ 1, j0n);   // async, overlaps with this step's WMMAs
            g2l_wait<4>();
            // WGP-scope prefetch of next V tile (consumed sync at step tail)
            __builtin_prefetch(&vb[headbase + (size_t)(j0n + (tid >> 1)) * DDIM], 0, 3);
        } else {
            g2l_wait<0>();
        }
        __syncthreads();

        // S = Q (16x64) x K^T (64x64): 4 key-tiles x 2 K-steps
        v8f s[4] = {};
#pragma unroll
        for (int jt = 0; jt < 4; ++jt) {
            v16bf bk0 = ld_frag(&Klds[buf][jt * 16 + ln][lh * 16]);
            s[jt] = wmma_bf16(qa0, bk0, s[jt]);
            v16bf bk1 = ld_frag(&Klds[buf][jt * 16 + ln][32 + lh * 16]);
            s[jt] = wmma_bf16(qa1, bk1, s[jt]);
        }

        // online softmax (per C-fragment row; N spread over 16-lane half)
#pragma unroll
        for (int r = 0; r < 8; ++r) {
            float p[4];
            float mx = -1e30f;
#pragma unroll
            for (int jt = 0; jt < 4; ++jt) { p[jt] = s[jt][r] * 0.125f; mx = fmaxf(mx, p[jt]); }
            mx = fmaxf(mx, __shfl_xor(mx, 1, 32));
            mx = fmaxf(mx, __shfl_xor(mx, 2, 32));
            mx = fmaxf(mx, __shfl_xor(mx, 4, 32));
            mx = fmaxf(mx, __shfl_xor(mx, 8, 32));
            float mnew = fmaxf(m_run[r], mx);
            float corr = __expf(m_run[r] - mnew);
            float rs = 0.0f;
#pragma unroll
            for (int jt = 0; jt < 4; ++jt) { p[jt] = __expf(p[jt] - mnew); rs += p[jt]; }
            rs += __shfl_xor(rs, 1, 32);
            rs += __shfl_xor(rs, 2, 32);
            rs += __shfl_xor(rs, 4, 32);
            rs += __shfl_xor(rs, 8, 32);
            l_run[r] = l_run[r] * corr + rs;
            m_run[r] = mnew;
#pragma unroll
            for (int t = 0; t < 4; ++t) o[t][r] *= corr;
            int prow = r + lh * 8;
#pragma unroll
            for (int jt = 0; jt < 4; ++jt)
                Plds[wave][prow][jt * 16 + ln] = (bf16_t)p[jt];
        }
        __syncthreads();

        // O += P (16x64) x V (64x64)
        v16bf pa0 = ld_frag(&Plds[wave][ln][lh * 16]);
        v16bf pa1 = ld_frag(&Plds[wave][ln][32 + lh * 16]);
#pragma unroll
        for (int t = 0; t < 4; ++t) {
            v16bf bv0 = ld_frag(&VldsT[buf][t * 16 + ln][lh * 16]);
            o[t] = wmma_bf16(pa0, bv0, o[t]);
            v16bf bv1 = ld_frag(&VldsT[buf][t * 16 + ln][32 + lh * 16]);
            o[t] = wmma_bf16(pa1, bv1, o[t]);
        }

        if (hn) stageV(buf ^ 1, j0n);   // behind the WMMAs; L2-warm
        __syncthreads();
    };

    stageK(0, 0);
    stageV(0, 0);
    for (int it = 0; it < TDIM / 64; it += 2) {
        jstep(0, (it + 1) * 64, true);                        // it+1 < 32 always
        jstep(1, (it + 2) * 64, (it + 2) < (TDIM / 64));
    }

    // normalize and store y in [b][t][h*64+d] bf16 for the output projection
#pragma unroll
    for (int r = 0; r < 8; ++r) {
        int trow = q0 + wave * 16 + r + lh * 8;
        float inv = 1.0f / l_run[r];
        size_t base = ((size_t)b_ * TDIM + trow) * CDIM + h * DDIM;
#pragma unroll
        for (int t = 0; t < 4; ++t)
            yb[base + t * 16 + ln] = (bf16_t)(o[t][r] * inv);
    }
}

// ---------------------------------------------------------------------------
// Output projection: out = y @ Wo^T, fp32 result.
// Block: 256 threads (8 waves). Block tile: 256 x 64, wave tile 32 x 64.
// Same two-phase pipeline. grid = (32, 16)
// ---------------------------------------------------------------------------
__global__ void __launch_bounds__(256, 2)
k_oproj(const bf16_t* __restrict__ yb, const bf16_t* __restrict__ wo,
        float* __restrict__ out) {
    __shared__ alignas(32) bf16_t Alds[2][256][32];
    __shared__ alignas(32) bf16_t Blds[2][64][32];

    const int m0 = blockIdx.x * 256;
    const int n0 = blockIdx.y * 64;

    const int tid  = threadIdx.x;
    const int wave = tid >> 5;
    const int lane = tid & 31;
    const int ln   = lane & 15;
    const int lh   = lane >> 4;

    v8f acc[2][4] = {};

    auto stage = [&](int buf, int k0) {
#pragma unroll
        for (int q = 0; q < 4; ++q) {
            int chunk = tid + q * 256;
            int row = chunk >> 2, off = (chunk & 3) * 8;
            g2l_copy16(&yb[(size_t)(m0 + row) * CDIM + k0 + off],
                       &Alds[buf][row][off]);
        }
        int brow = tid >> 2, boff = (tid & 3) * 8;
        g2l_copy16(&wo[(size_t)(n0 + brow) * CDIM + k0 + boff],
                   &Blds[buf][brow][boff]);
    };

    auto kstep = [&](int buf, int k0n, bool hn) {
        if (hn) {
            stage(buf ^ 1, k0n);
            g2l_wait<5>();
        } else {
            g2l_wait<0>();
        }
        __syncthreads();
        v16bf a0 = ld_frag(&Alds[buf][wave * 32 + ln][lh * 16]);
        v16bf a1 = ld_frag(&Alds[buf][wave * 32 + 16 + ln][lh * 16]);
#pragma unroll
        for (int t = 0; t < 4; ++t) {
            v16bf b = ld_frag(&Blds[buf][t * 16 + ln][lh * 16]);
            acc[0][t] = wmma_bf16(a0, b, acc[0][t]);
            acc[1][t] = wmma_bf16(a1, b, acc[1][t]);
        }
        __syncthreads();
    };

    stage(0, 0);
    for (int it = 0; it < CDIM / 32; it += 2) {
        kstep(0, (it + 1) * 32, true);
        kstep(1, (it + 2) * 32, (it + 2) < (CDIM / 32));
    }

#pragma unroll
    for (int u = 0; u < 2; ++u) {
        const int rbase = m0 + wave * 32 + u * 16;
#pragma unroll
        for (int r = 0; r < 8; ++r) {
            int mrow = rbase + r + lh * 8;
#pragma unroll
            for (int t = 0; t < 4; ++t)
                out[(size_t)mrow * CDIM + n0 + t * 16 + ln] = acc[u][t][r];
        }
    }
}

// ---------------------------------------------------------------------------
extern "C" void kernel_launch(void* const* d_in, const int* in_sizes, int n_in,
                              void* d_out, int out_size, void* d_ws, size_t ws_size,
                              hipStream_t stream) {
    const float* x    = (const float*)d_in[0];
    const float* cosp = (const float*)d_in[1];
    const float* sinp = (const float*)d_in[2];
    const float* Wq   = (const float*)d_in[3];
    const float* Wk   = (const float*)d_in[4];
    const float* Wv   = (const float*)d_in[5];
    const float* Wo   = (const float*)d_in[6];

    char* ws = (char*)d_ws;
    const size_t MB = 1ull << 20;
    bf16_t* xb  = (bf16_t*)(ws);            // 16 MB
    bf16_t* wqb = (bf16_t*)(ws + 16 * MB);  //  2 MB
    bf16_t* wkb = (bf16_t*)(ws + 18 * MB);
    bf16_t* wvb = (bf16_t*)(ws + 20 * MB);
    bf16_t* wob = (bf16_t*)(ws + 22 * MB);
    bf16_t* qb  = (bf16_t*)(ws + 24 * MB);  // 16 MB, [b][h][t][d]
    bf16_t* kb  = (bf16_t*)(ws + 40 * MB);
    bf16_t* vb  = (bf16_t*)(ws + 56 * MB);
    bf16_t* yb  = (bf16_t*)(ws + 72 * MB);  // 16 MB, [b][t][h*64+d]

    const int nx = BDIM * TDIM * CDIM;   // 8,388,608
    const int nw = CDIM * CDIM;          // 1,048,576

    k_f32_to_bf16<<<nx / 1024, 256, 0, stream>>>(x,  xb,  nx);
    k_f32_to_bf16<<<nw / 1024, 256, 0, stream>>>(Wq, wqb, nw);
    k_f32_to_bf16<<<nw / 1024, 256, 0, stream>>>(Wk, wkb, nw);
    k_f32_to_bf16<<<nw / 1024, 256, 0, stream>>>(Wv, wvb, nw);
    k_f32_to_bf16<<<nw / 1024, 256, 0, stream>>>(Wo, wob, nw);

    dim3 g1(BDIM * TDIM / 256, HDIM, 3);
    k_qkv<<<g1, 256, 0, stream>>>(xb, wqb, wkb, wvb, cosp, sinp, qb, kb, vb);

    dim3 g2(TDIM / 64, BDIM * HDIM);
    k_attn<<<g2, 128, 0, stream>>>(qb, kb, vb, yb);

    dim3 g3(BDIM * TDIM / 256, CDIM / 64);
    k_oproj<<<g3, 256, 0, stream>>>(yb, wob, (float*)d_out);
}
